// SelfAttention_69363721830412
// MI455X (gfx1250) — matrix operand order repaired and checked by reading, not compiled, not run on hardware
//
#include <hip/hip_runtime.h>

#define SEQ   2048
#define NH    16
#define HD    64
#define BATCH 4

typedef _Float16 h2   __attribute__((ext_vector_type(2)));
typedef _Float16 h4   __attribute__((ext_vector_type(4)));
typedef _Float16 h8   __attribute__((ext_vector_type(8)));
typedef _Float16 v16h __attribute__((ext_vector_type(16)));
typedef float    v8f  __attribute__((ext_vector_type(8)));

__device__ __forceinline__ v16h pack16(h8 lo, h8 hi) {
  v16h r;
#pragma unroll
  for (int i = 0; i < 8; ++i) { r[i] = lo[i]; r[i + 8] = hi[i]; }
  return r;
}

__device__ __forceinline__ v16h load16h(const _Float16* p0, const _Float16* p1) {
  return pack16(*(const h8*)p0, *(const h8*)p1);
}

// Load 2x8 contiguous f32 and convert to the 16-half WMMA operand.
__device__ __forceinline__ v16h cvt16f(const float* p0, const float* p1) {
  v8f a = *(const v8f*)p0;
  v8f b = *(const v8f*)p1;
  v16h r;
#pragma unroll
  for (int i = 0; i < 8; ++i) { r[i] = (_Float16)a[i]; r[i + 8] = (_Float16)b[i]; }
  return r;
}

__device__ __forceinline__ v8f wmma16(v16h a, v16h b, v8f c) {
  return __builtin_amdgcn_wmma_f32_16x16x32_f16(false, a, false, b, (short)0, c,
                                                false, false);
}

// CDNA5 async DMA: 16B global -> LDS, tracked by ASYNCcnt (no VGPR pass).
// Generic pointers to __shared__ keep the LDS byte offset in addr[31:0].
__device__ __forceinline__ void async_copy16(const _Float16* g,
                                             const _Float16* l) {
  const unsigned lo = (unsigned)(unsigned long long)l;
  const unsigned long long ga = (unsigned long long)g;
  asm volatile("global_load_async_to_lds_b128 %0, %1, off"
               :: "v"(lo), "v"(ga) : "memory");
}

__device__ __forceinline__ void wait_async0() {
  asm volatile("s_wait_asynccnt 0x0" ::: "memory");
}

// ---------------------------------------------------------------------------
// Kernel 0: one-shot f32 -> f16 conversion for shared weight matrices.
// ---------------------------------------------------------------------------
__global__ __launch_bounds__(256) void cvt_f16_kernel(
    const float* __restrict__ in, _Float16* __restrict__ out, int n) {
  const int i = blockIdx.x * 256 + threadIdx.x;
  if (i < n) out[i] = (_Float16)in[i];
}

// ---------------------------------------------------------------------------
// Kernel 1: per-head linear projection  out = (x @ W^T + b) * outScale, f16 out
// One wave = one (b,s): 16 rows (the 16 heads) x 64 cols.  K = 64 (2 wmma steps)
// ---------------------------------------------------------------------------
__global__ __launch_bounds__(256) void qkv_proj_kernel(
    const float* __restrict__ X, const _Float16* __restrict__ Wh,
    const float* __restrict__ bias, _Float16* __restrict__ Out, float outScale) {
  const int w    = threadIdx.x >> 5;
  const int lane = threadIdx.x & 31;
  const int lp   = lane & 15;
  const int hi   = lane >> 4;
  const int ka   = hi * 8;

  const int mtile = blockIdx.x * 8 + w;    // == b*SEQ + s  (16 rows = 16 heads)
  const int b = mtile >> 11;
  const int s = mtile & 2047;

  const float* xr = X + (size_t)(mtile * 16 + lp) * HD;

  v8f acc[4] = {};
#pragma unroll
  for (int kb = 0; kb < HD; kb += 32) {
    const v16h a = cvt16f(xr + kb + hi * 16, xr + kb + hi * 16 + 8);
#pragma unroll
    for (int nt = 0; nt < 4; ++nt) {
      const _Float16* wr = Wh + (size_t)(nt * 16 + lp) * HD + kb + hi * 16;
      acc[nt] = wmma16(a, load16h(wr, wr + 8), acc[nt]);
    }
  }

#pragma unroll
  for (int nt = 0; nt < 4; ++nt) {
    const int col = nt * 16 + lp;
    const float bv = bias[col];
#pragma unroll
    for (int v = 0; v < 8; ++v) {
      const int h = hi * 8 + v;            // C-layout row = head index
      const size_t orow = (size_t)(b * NH + h) * SEQ + s;
      Out[orow * HD + col] = (_Float16)((acc[nt][v] + bv) * outScale);
    }
  }
}

// ---------------------------------------------------------------------------
// Kernel 2: flash attention.  Block = 4 waves = 64 query rows of one (b,h).
// KV chunk = 64 keys, double-buffered:
//   - K tile DMA'd straight to LDS with global_load_async_to_lds_b128
//     (issued right after the barrier, waited with s_wait_asynccnt before the
//      next barrier publishes the buffer)
//   - V tile staged through VGPRs (needs transpose), packed b64 LDS stores
// One __syncthreads per chunk; P bounced through a per-wave LDS tile to turn
// the WMMA C layout into the A layout for the two K=32 P@V steps.
// ---------------------------------------------------------------------------
#define KSTRIDE 72     // 64x64 K tile rows padded to 144B (16B aligned)
#define VSTRIDE 72     // 64(dims) x 64(keys) V^T tile, padded
#define PSTRIDE 72     // 16x64 P tile per wave, padded
#define NCHUNK  (SEQ / 64)

__global__ __launch_bounds__(128) void flash_attn_kernel(
    const _Float16* __restrict__ Qp, const _Float16* __restrict__ Kp,
    const _Float16* __restrict__ Vp, _Float16* __restrict__ ctx) {
  __shared__ _Float16 Ksh[2][64 * KSTRIDE];   // 2 x 9216 B
  __shared__ _Float16 Vts[2][HD * VSTRIDE];   // 2 x 9216 B
  __shared__ _Float16 Psh[4][16 * PSTRIDE];   // 4 x 2304 B   (total 46080 B)

  const int tid  = threadIdx.x;
  const int w    = tid >> 5;
  const int lane = tid & 31;
  const int lp   = lane & 15;
  const int hi   = lane >> 4;
  const int ka   = hi * 8;

  const int bh   = blockIdx.x;
  const int b    = bh >> 4;
  const int h    = bh & 15;
  const int qblk = blockIdx.y * 64;

  // Q rows live in registers for the whole KV sweep (1/sqrt(d) pre-folded).
  const _Float16* qr = Qp + ((size_t)bh * SEQ + qblk + w * 16 + lp) * HD;
  const v16h aq0 = load16h(qr + hi * 16,      qr + hi * 16 + 8);
  const v16h aq1 = load16h(qr + 32 + hi * 16, qr + 40 + hi * 16);

  v8f oacc[4] = {};
  float mrow[8], lrow[8];
#pragma unroll
  for (int v = 0; v < 8; ++v) { mrow[v] = -3.0e38f; lrow[v] = 0.0f; }

  // K async mapping: thread owns key row (tid>>1), 32-half segment (tid&1)*32
  const int akrow = tid >> 1;
  const int aseg  = (tid & 1) * 32;
  const size_t gkbase = (size_t)bh * SEQ * HD + (size_t)akrow * HD + aseg;

  // V staging mapping: thread owns 4 key rows x 8-wide d segment
  const int vkr = (tid & 15) * 4;        // 0,4,...,60
  const int vds = (tid >> 4) * 8;        // 0,8,...,56
  const size_t gvbase = (size_t)bh * SEQ * HD + (size_t)vkr * HD + vds;

  h8 sv[4];

#define ASYNC_K(kvoff, buf)                                                   \
  {                                                                           \
    const _Float16* g = Kp + gkbase + (size_t)(kvoff)*HD;                     \
    _Float16* l = &Ksh[buf][akrow * KSTRIDE + aseg];                          \
    async_copy16(g, l);           async_copy16(g + 8, l + 8);                 \
    async_copy16(g + 16, l + 16); async_copy16(g + 24, l + 24);               \
  }

#define STAGE_LOAD_V(kvoff)                                                   \
  {                                                                           \
    const _Float16* gv = Vp + gvbase + (size_t)(kvoff)*HD;                    \
    sv[0] = *(const h8*)gv;            sv[1] = *(const h8*)(gv + HD);         \
    sv[2] = *(const h8*)(gv + 2 * HD); sv[3] = *(const h8*)(gv + 3 * HD);     \
  }

#define STAGE_STORE_V(buf)                                                    \
  {                                                                           \
    _Float16* vt = &Vts[buf][0];                                              \
    _Pragma("unroll")                                                         \
    for (int j = 0; j < 8; ++j) {                                             \
      h4 t;                                                                   \
      t[0] = sv[0][j]; t[1] = sv[1][j]; t[2] = sv[2][j]; t[3] = sv[3][j];     \
      *(h4*)&vt[(vds + j) * VSTRIDE + vkr] = t;                               \
    }                                                                         \
  }

  ASYNC_K(0, 0);
  STAGE_LOAD_V(0);
  STAGE_STORE_V(0);
  wait_async0();

#pragma unroll 1
  for (int c = 0; c < NCHUNK; ++c) {
    const int cur = c & 1;
    __syncthreads();   // publish buf `cur`; all waves done reading buf cur^1
    if (c + 1 < NCHUNK) {
      ASYNC_K((c + 1) * 64, cur ^ 1);    // DMA next K tile during compute
      STAGE_LOAD_V((c + 1) * 64);        // next V tile into registers
    }

    // ---- scores: 4 key tiles x (K=64 as 2 wmma steps) -------------------
    v8f s[4];
#pragma unroll
    for (int nt = 0; nt < 4; ++nt) {
      const _Float16* krow = &Ksh[cur][(nt * 16 + lp) * KSTRIDE + hi * 16];
      v8f z = {};
      z = wmma16(aq0, load16h(krow, krow + 8), z);
      s[nt] = wmma16(aq1, load16h(krow + 32, krow + 40), z);
    }

    // ---- online softmax (row reductions inside 16-lane halves) ----------
    _Float16* pw = &Psh[w][0];
#pragma unroll
    for (int v = 0; v < 8; ++v) {
      float mx = fmaxf(fmaxf(s[0][v], s[1][v]), fmaxf(s[2][v], s[3][v]));
#pragma unroll
      for (int off = 1; off < 16; off <<= 1)
        mx = fmaxf(mx, __shfl_xor(mx, off, 32));
      const float mnew = fmaxf(mrow[v], mx);
      const float alpha = __expf(mrow[v] - mnew);
      mrow[v] = mnew;
      const int r = (hi * 8 + v) * PSTRIDE + lp;
      float rs = 0.0f;
#pragma unroll
      for (int nt = 0; nt < 4; ++nt) {
        const float p = __expf(s[nt][v] - mnew);
        rs += p;
        pw[r + nt * 16] = (_Float16)p;     // P -> per-wave LDS tile (C layout)
      }
#pragma unroll
      for (int off = 1; off < 16; off <<= 1)
        rs += __shfl_xor(rs, off, 32);
      lrow[v] = lrow[v] * alpha + rs;
#pragma unroll
      for (int nt = 0; nt < 4; ++nt) oacc[nt][v] *= alpha;
    }

    // ---- reload P in A layout (same-wave DS ops are in order) -----------
    const _Float16* prd = &Psh[w][lp * PSTRIDE];
    const v16h pa0 = load16h(prd + hi * 16,      prd + hi * 16 + 8);   // keys  0..31
    const v16h pa1 = load16h(prd + 32 + hi * 16, prd + 40 + hi * 16);   // keys 32..63

    // ---- O += P(16x64) @ V(64x64): B[k][n] = Vts[n][k] contiguous -------
#pragma unroll
    for (int nt = 0; nt < 4; ++nt) {
      const _Float16* vr = &Vts[cur][(nt * 16 + lp) * VSTRIDE + hi * 16];
      oacc[nt] = wmma16(pa0, load16h(vr, vr + 8), oacc[nt]);
      oacc[nt] = wmma16(pa1, load16h(vr + 32, vr + 40), oacc[nt]);
    }

    if (c + 1 < NCHUNK) {
      STAGE_STORE_V(cur ^ 1);   // safe: all waves past this iteration's barrier
      wait_async0();            // K DMA durable in LDS before next barrier
    }
  }

  // normalize and scatter to merged-head ctx [B, S, 1024] (f16)
#pragma unroll
  for (int nt = 0; nt < 4; ++nt) {
#pragma unroll
    for (int v = 0; v < 8; ++v) {
      const int sq = qblk + w * 16 + hi * 8 + v;
      ctx[((size_t)b * SEQ + sq) * 1024 + h * HD + nt * 16 + lp] =
          (_Float16)(oacc[nt][v] / lrow[v]);
    }
  }
}

// ---------------------------------------------------------------------------
// Kernel 3: out = ctx @ Wp^T + bp (f32 out).  Wave per (16-row, 64-col) slab,
// K = 1024.  Both operands pre-converted f16; Wp16 (2MB) lives in L2.
// ---------------------------------------------------------------------------
__global__ __launch_bounds__(256) void out_proj_kernel(
    const _Float16* __restrict__ ctx, const _Float16* __restrict__ Wp,
    const float* __restrict__ bp, float* __restrict__ out) {
  const int w    = threadIdx.x >> 5;
  const int lane = threadIdx.x & 31;
  const int lp   = lane & 15;
  const int hi   = lane >> 4;
  const int ka   = hi * 8;

  const int g     = blockIdx.x * 8 + w;   // 8192 waves total
  const int mtile = g >> 4;               // 512 row tiles
  const int nslab = g & 15;               // 16 column slabs of 64

  const _Float16* ar = ctx + (size_t)(mtile * 16 + lp) * 1024;

  v8f acc[4] = {};
#pragma unroll 1
  for (int kb = 0; kb < 1024; kb += 32) {
    const v16h a = load16h(ar + kb + hi * 16, ar + kb + hi * 16 + 8);
#pragma unroll
    for (int nt = 0; nt < 4; ++nt) {
      const _Float16* wr =
          Wp + (size_t)(nslab * 64 + nt * 16 + lp) * 1024 + kb + hi * 16;
      acc[nt] = wmma16(a, load16h(wr, wr + 8), acc[nt]);
    }
  }

#pragma unroll
  for (int nt = 0; nt < 4; ++nt) {
    const int col = nslab * 64 + nt * 16 + lp;
    const float bv = bp[col];
#pragma unroll
    for (int v = 0; v < 8; ++v) {
      out[(size_t)(mtile * 16 + hi * 8 + v) * 1024 + col] = acc[nt][v] + bv;
    }
  }
}

// ---------------------------------------------------------------------------
extern "C" void kernel_launch(void* const* d_in, const int* in_sizes, int n_in,
                              void* d_out, int out_size, void* d_ws,
                              size_t ws_size, hipStream_t stream) {
  (void)in_sizes; (void)n_in; (void)out_size; (void)ws_size;
  const float* q  = (const float*)d_in[0];
  const float* k  = (const float*)d_in[1];
  const float* v  = (const float*)d_in[2];
  const float* Wq = (const float*)d_in[3];
  const float* bq = (const float*)d_in[4];
  const float* Wk = (const float*)d_in[5];
  const float* bk = (const float*)d_in[6];
  const float* Wv = (const float*)d_in[7];
  const float* bv = (const float*)d_in[8];
  const float* Wp = (const float*)d_in[9];
  const float* bp = (const float*)d_in[10];

  const size_t elems = (size_t)BATCH * NH * SEQ * HD;  // 8,388,608
  _Float16* Qp    = (_Float16*)d_ws;      // 16 MB
  _Float16* Kp    = Qp + elems;           // 16 MB
  _Float16* Vp    = Kp + elems;           // 16 MB
  _Float16* ctx16 = Vp + elems;           // 16 MB
  _Float16* Wp16  = ctx16 + elems;        //  2 MB
  _Float16* Wq16  = Wp16 + (size_t)1024 * 1024;
  _Float16* Wk16  = Wq16 + HD * HD;
  _Float16* Wv16  = Wk16 + HD * HD;       // total ~66 MB of ws

  // one-shot weight conversions (shared operands -> f16)
  cvt_f16_kernel<<<dim3(16), 256, 0, stream>>>(Wq, Wq16, HD * HD);
  cvt_f16_kernel<<<dim3(16), 256, 0, stream>>>(Wk, Wk16, HD * HD);
  cvt_f16_kernel<<<dim3(16), 256, 0, stream>>>(Wv, Wv16, HD * HD);
  cvt_f16_kernel<<<dim3(4096), 256, 0, stream>>>(Wp, Wp16, 1024 * 1024);

  // fold 1/sqrt(HEAD_DIM)=0.125 into stored Q
  qkv_proj_kernel<<<dim3(1024), 256, 0, stream>>>(q, Wq16, bq, Qp, 0.125f);
  qkv_proj_kernel<<<dim3(1024), 256, 0, stream>>>(k, Wk16, bk, Kp, 1.0f);
  qkv_proj_kernel<<<dim3(1024), 256, 0, stream>>>(v, Wv16, bv, Vp, 1.0f);
  flash_attn_kernel<<<dim3(BATCH * NH, SEQ / 64), 128, 0, stream>>>(Qp, Kp, Vp,
                                                                    ctx16);
  out_proj_kernel<<<dim3(1024), 256, 0, stream>>>(ctx16, Wp16, bp,
                                                  (float*)d_out);
}
